// AdvancedAutoInformerModel_40235253629093
// MI455X (gfx1250) — matrix operand order repaired and checked by reading, not compiled, not run on hardware
//
#include <hip/hip_runtime.h>
#include <hip/hip_bf16.h>
#include <math.h>
#include <stdint.h>

typedef __bf16 bf16_t;
typedef __attribute__((ext_vector_type(16))) __bf16 v16bf;
typedef __attribute__((ext_vector_type(8)))  __bf16 v8bf;
typedef __attribute__((ext_vector_type(8)))  float  v8f;
typedef __attribute__((ext_vector_type(4)))  int    v4i;
typedef __attribute__((ext_vector_type(8)))  int    v8i;

namespace {
constexpr int  kB   = 32;
constexpr int  kT   = 4000;
constexpr int  kC   = 16;
constexpr int  kD   = 256;
constexpr int  kF   = 1024;
constexpr int  kL   = 4;
constexpr int  kBlk = 20;
constexpr int  kNB  = kT / kBlk;      // 200 blocks per sequence
constexpr int  kH   = 8;
constexpr int  kHD  = kD / kH;        // 32
constexpr long kBT  = (long)kB * kT;  // 128000 tokens (divisible by 64)
constexpr int  kBStride = 72;         // padded LDS row stride (bf16 elems)
}

// ---------------------------------------------------------------------------
// WMMA fragment helpers (CDNA5 16x16x32 bf16, wave32)
// A 16x32: lane L -> row L&15; 16 contiguous K starting at kbase,
//          kbase = 16*(L>=16).
// B 32x16: lane L -> col L&15; 16 contiguous K starting at 16*(L>=16).
// C/D:     elem e -> (M = e + 8*(L>=16), N = L&15).
// ---------------------------------------------------------------------------
__device__ __forceinline__ v16bf frag_a_load(const bf16_t* row, int k0) {
  const v8bf* p = (const v8bf*)(row + k0);
  v8bf lo = p[0];
  v8bf hi = p[1];  // +8 elements
  return __builtin_shufflevector(lo, hi, 0, 1, 2, 3, 4, 5, 6, 7,
                                 8, 9, 10, 11, 12, 13, 14, 15);
}

// B fragment from LDS: 32 contiguous bytes as two 16-B loads (row stride is
// 16-B aligned but not 32-B aligned due to the 72-element padding).
__device__ __forceinline__ v16bf frag_b_load_lds(const bf16_t* rowp, int k0) {
  const v8bf* p = (const v8bf*)(rowp + k0);
  v8bf lo = p[0];
  v8bf hi = p[1];
  return __builtin_shufflevector(lo, hi, 0, 1, 2, 3, 4, 5, 6, 7,
                                 8, 9, 10, 11, 12, 13, 14, 15);
}

__device__ __forceinline__ v16bf frag_b_load(const bf16_t* row, int k0) {
  return *(const v16bf*)(row + k0);
}

__device__ __forceinline__ v8f wmma_bf16(v16bf a, v16bf b, v8f c) {
  return __builtin_amdgcn_wmma_f32_16x16x32_bf16(false, a, false, b,
                                                 (short)0, c, false, false);
}

// ---------------------------------------------------------------------------
// Tensor Data Mover: load one 64(N-rows) x 64(K-elems) bf16 tile from global
// (row stride = row_stride_elems) into LDS at lds_addr, padding each 128-B
// row with 16 B so LDS row stride is 72 elements (bank-conflict break).
// D# layout per CDNA5 ISA 8.3/8.4. Issued by one wave; TENSORcnt-tracked.
// ---------------------------------------------------------------------------
__device__ __forceinline__ void tdm_load_tile_64x64(uint32_t lds_addr,
                                                    const bf16_t* gsrc,
                                                    int row_stride_elems) {
  const uint64_t ga = (uint64_t)(uintptr_t)gsrc;
  v4i g0;
  g0[0] = 1;                                    // count=1, user descriptor
  g0[1] = (int)lds_addr;                        // lds_addr (bytes)
  g0[2] = (int)(uint32_t)ga;                    // global_addr[31:0]
  g0[3] = (int)(uint32_t)(((ga >> 32) & 0x01FFFFFFull) | 0x80000000ull);
                                                // global_addr[56:32] | type=2
  v8i g1;
  g1[0] = (1 << 16)     // data_size = 2 bytes
        | (1 << 20)     // pad_enable
        | (4 << 22)     // pad_interval: 32 DWORDs (=128 B = one 64-elem row)
        | (3 << 25);    // pad_amount: 4 DWORDs (=16 B = 8 bf16 elems)
  g1[1] = 0;                      // tensor_dim0[15:0] (dim0 = 1<<30)
  g1[2] = 0x4000;                 // tensor_dim0[31:16]; tensor_dim1[15:0]=0
  g1[3] = 0x4000 | (64 << 16);    // tensor_dim1[31:16]; tile_dim0 = 64
  g1[4] = 64;                     // tile_dim1 = 64; tile_dim2 = 0
  g1[5] = row_stride_elems;       // tensor_dim0_stride[31:0]
  g1[6] = 0;                      // stride0[47:32]=0; stride1[15:0]=0
  g1[7] = 0;
  asm volatile("tensor_load_to_lds %0, %1" :: "s"(g0), "s"(g1) : "memory");
}

__device__ __forceinline__ void epilogue(v8f acc, int mbase, int n, int N,
                                         const float* bias, const float* res,
                                         float* out32, bf16_t* out16, int act) {
#pragma unroll
  for (int e = 0; e < 8; ++e) {
    long off = (long)(mbase + e) * N + n;
    float v = acc[e] + bias[n];
    if (res) v += res[off];
    if (act) v = fmaxf(v, 0.0f);
    if (out32) out32[off] = v;
    if (out16) out16[off] = (bf16_t)v;
  }
}

// ---------------------------------------------------------------------------
// Generic GEMM: Y[M,N] = act(X[M,K] @ W[N,K]^T + bias) (+ res), X/W bf16.
// grid = (M/64, N/64), block = 128 (4 waves). Each wave computes a 16x64
// tile; the 64x64 weight tile is TDM-staged into double-buffered LDS and
// shared by all 4 waves. A is streamed from global (b128 + prefetch).
// ---------------------------------------------------------------------------
__global__ __launch_bounds__(128)
void gemm_bf16_wmma(const bf16_t* __restrict__ A, const bf16_t* __restrict__ W,
                    const float* __restrict__ bias, const float* __restrict__ res,
                    float* __restrict__ out32, bf16_t* __restrict__ out16,
                    int N, int K, int act) {
  __shared__ bf16_t sB[2][64 * kBStride];
  const int tid = threadIdx.x;
  const int wave = tid >> 5;
  const int lane = tid & 31;
  const int tm = blockIdx.x, tn = blockIdx.y;
  const int half = lane >> 4;
  const int ka = half << 3, kb = half << 4;
  const int n_lane = lane & 15;
  const bf16_t* a_row = A + (long)(tm * 64 + wave * 16 + n_lane) * K;
  const bf16_t* wtile = W + (long)(tn * 64) * K;
  const int nchunks = K >> 6;

  if (wave == 0) {
    tdm_load_tile_64x64((uint32_t)(uintptr_t)&sB[0][0], wtile, K);
  }

  v8f c0 = {}, c1 = {}, c2 = {}, c3 = {};
  for (int c = 0; c < nchunks; ++c) {
    if (wave == 0) {
      if (c + 1 < nchunks) {
        tdm_load_tile_64x64((uint32_t)(uintptr_t)&sB[(c + 1) & 1][0],
                            wtile + (c + 1) * 64, K);
        __builtin_amdgcn_s_wait_tensorcnt(1);  // chunk c resident
      } else {
        __builtin_amdgcn_s_wait_tensorcnt(0);
      }
    }
    __syncthreads();  // LDS chunk c visible to all waves
    const bf16_t* bb = &sB[c & 1][0];
    const int kk0 = c * 64;
#pragma unroll
    for (int k2 = 0; k2 < 64; k2 += 32) {
      __builtin_prefetch(a_row + kk0 + k2 + 128, 0, 1);
      v16bf a = frag_a_load(a_row, kk0 + k2 + kb);
      v16bf b0 = frag_b_load_lds(bb + (0 + n_lane) * kBStride, k2 + kb);
      v16bf b1 = frag_b_load_lds(bb + (16 + n_lane) * kBStride, k2 + kb);
      v16bf b2 = frag_b_load_lds(bb + (32 + n_lane) * kBStride, k2 + kb);
      v16bf b3 = frag_b_load_lds(bb + (48 + n_lane) * kBStride, k2 + kb);
      c0 = wmma_bf16(a, b0, c0);
      c1 = wmma_bf16(a, b1, c1);
      c2 = wmma_bf16(a, b2, c2);
      c3 = wmma_bf16(a, b3, c3);
    }
    __syncthreads();  // all waves done with chunk c before TDM refill
  }
  const int mbase = tm * 64 + wave * 16 + (half << 3);
  epilogue(c0, mbase, tn * 64 + 0 + n_lane, N, bias, res, out32, out16, act);
  epilogue(c1, mbase, tn * 64 + 16 + n_lane, N, bias, res, out32, out16, act);
  epilogue(c2, mbase, tn * 64 + 32 + n_lane, N, bias, res, out32, out16, act);
  epilogue(c3, mbase, tn * 64 + 48 + n_lane, N, bias, res, out32, out16, act);
}

// ---------------------------------------------------------------------------
// conv2 (D=256 -> D=256, k=3, SAME) + ReLU as WMMA GEMM over virtual K=768.
// W pre-permuted to [o][tap*256 + i] bf16. grid = (BT/16, 4), block 32.
// ---------------------------------------------------------------------------
__global__ __launch_bounds__(32)
void conv2_relu_wmma(const bf16_t* __restrict__ h1, const bf16_t* __restrict__ W,
                     const float* __restrict__ bias, float* __restrict__ out32) {
  const int lane = threadIdx.x;
  const int tm = blockIdx.x, tn = blockIdx.y;
  const int half = lane >> 4;
  const int ka = half << 3, kb = half << 4;
  const int n_lane = lane & 15;
  const long g0 = (long)tm * 16;
  const int b = (int)(g0 / kT);
  const int t = (int)(g0 % kT) + n_lane;   // A-fragment row token
  const bf16_t* w0 = W + (long)(tn * 64 + 0 + n_lane) * 768;
  const bf16_t* w1 = W + (long)(tn * 64 + 16 + n_lane) * 768;
  const bf16_t* w2 = W + (long)(tn * 64 + 32 + n_lane) * 768;
  const bf16_t* w3 = W + (long)(tn * 64 + 48 + n_lane) * 768;
  v8f c0 = {}, c1 = {}, c2 = {}, c3 = {};
  for (int kk = 0; kk < 768; kk += 32) {
    const int tap = kk >> 8;          // 0..2, K-step never crosses a tap
    const int tp = t + tap - 1;
    v16bf a;
    if (tp < 0 || tp >= kT) {
#pragma unroll
      for (int e = 0; e < 16; ++e) a[e] = (bf16_t)0.0f;
    } else {
      const bf16_t* row = h1 + ((long)b * kT + tp) * kD;
      a = frag_a_load(row, (kk - (tap << 8)) + kb);
    }
    c0 = wmma_bf16(a, frag_b_load(w0, kk + kb), c0);
    c1 = wmma_bf16(a, frag_b_load(w1, kk + kb), c1);
    c2 = wmma_bf16(a, frag_b_load(w2, kk + kb), c2);
    c3 = wmma_bf16(a, frag_b_load(w3, kk + kb), c3);
  }
  const int mbase = tm * 16 + (half << 3);
  epilogue(c0, mbase, tn * 64 + 0 + n_lane, kD, bias, nullptr, out32, nullptr, 1);
  epilogue(c1, mbase, tn * 64 + 16 + n_lane, kD, bias, nullptr, out32, nullptr, 1);
  epilogue(c2, mbase, tn * 64 + 32 + n_lane, kD, bias, nullptr, out32, nullptr, 1);
  epilogue(c3, mbase, tn * 64 + 48 + n_lane, kD, bias, nullptr, out32, nullptr, 1);
}

// ---------------------------------------------------------------------------
// out = hres + (A@Wt^T + tb) + sin(A@Ws^T + sb). grid=(BT/16,4), block 32.
// ---------------------------------------------------------------------------
__device__ __forceinline__ void ts_epi(v8f at, v8f as, int mbase, int n,
                                       const float* tb, const float* sb,
                                       const float* hres, float* out32,
                                       bf16_t* out16) {
#pragma unroll
  for (int e = 0; e < 8; ++e) {
    long off = (long)(mbase + e) * kD + n;
    float v = hres[off] + (at[e] + tb[n]) + sinf(as[e] + sb[n]);
    out32[off] = v;
    out16[off] = (bf16_t)v;
  }
}

__global__ __launch_bounds__(32)
void trend_season_wmma(const bf16_t* __restrict__ A,
                       const bf16_t* __restrict__ Wt, const bf16_t* __restrict__ Ws,
                       const float* __restrict__ tb, const float* __restrict__ sb,
                       const float* __restrict__ hres,
                       float* __restrict__ out32, bf16_t* __restrict__ out16) {
  const int lane = threadIdx.x;
  const int tm = blockIdx.x, tn = blockIdx.y;
  const int half = lane >> 4;
  const int ka = half << 3, kb = half << 4;
  const int n_lane = lane & 15;
  const bf16_t* a_row = A + (long)(tm * 16 + n_lane) * kD;
  const bf16_t* wt0 = Wt + (long)(tn * 64 + 0 + n_lane) * kD;
  const bf16_t* wt1 = Wt + (long)(tn * 64 + 16 + n_lane) * kD;
  const bf16_t* wt2 = Wt + (long)(tn * 64 + 32 + n_lane) * kD;
  const bf16_t* wt3 = Wt + (long)(tn * 64 + 48 + n_lane) * kD;
  const bf16_t* ws0 = Ws + (long)(tn * 64 + 0 + n_lane) * kD;
  const bf16_t* ws1 = Ws + (long)(tn * 64 + 16 + n_lane) * kD;
  const bf16_t* ws2 = Ws + (long)(tn * 64 + 32 + n_lane) * kD;
  const bf16_t* ws3 = Ws + (long)(tn * 64 + 48 + n_lane) * kD;
  v8f t0 = {}, t1 = {}, t2 = {}, t3 = {};
  v8f s0 = {}, s1 = {}, s2 = {}, s3 = {};
  for (int kk = 0; kk < kD; kk += 32) {
    v16bf a = frag_a_load(a_row, kk + kb);
    t0 = wmma_bf16(a, frag_b_load(wt0, kk + kb), t0);
    t1 = wmma_bf16(a, frag_b_load(wt1, kk + kb), t1);
    t2 = wmma_bf16(a, frag_b_load(wt2, kk + kb), t2);
    t3 = wmma_bf16(a, frag_b_load(wt3, kk + kb), t3);
    s0 = wmma_bf16(a, frag_b_load(ws0, kk + kb), s0);
    s1 = wmma_bf16(a, frag_b_load(ws1, kk + kb), s1);
    s2 = wmma_bf16(a, frag_b_load(ws2, kk + kb), s2);
    s3 = wmma_bf16(a, frag_b_load(ws3, kk + kb), s3);
  }
  const int mbase = tm * 16 + (half << 3);
  ts_epi(t0, s0, mbase, tn * 64 + 0 + n_lane, tb, sb, hres, out32, out16);
  ts_epi(t1, s1, mbase, tn * 64 + 16 + n_lane, tb, sb, hres, out32, out16);
  ts_epi(t2, s2, mbase, tn * 64 + 32 + n_lane, tb, sb, hres, out32, out16);
  ts_epi(t3, s3, mbase, tn * 64 + 48 + n_lane, tb, sb, hres, out32, out16);
}

// ---------------------------------------------------------------------------
// LayerNorm over D=256; one wave per token; writes fp32 + bf16 copies.
// grid = BT/8, block = 256 (8 waves).
// ---------------------------------------------------------------------------
__global__ __launch_bounds__(256)
void layernorm_kernel(const float* __restrict__ in, const float* __restrict__ g,
                      const float* __restrict__ b, float* __restrict__ out32,
                      bf16_t* __restrict__ out16) {
  const int lane = threadIdx.x & 31;
  const int wid = threadIdx.x >> 5;
  const long tok = (long)blockIdx.x * 8 + wid;
  const float* row = in + tok * kD;
  float4 u = ((const float4*)row)[lane];
  float4 w = ((const float4*)row)[lane + 32];
  float s = u.x + u.y + u.z + u.w + w.x + w.y + w.z + w.w;
#pragma unroll
  for (int o = 16; o > 0; o >>= 1) s += __shfl_xor(s, o, 32);
  const float mean = s * (1.0f / 256.0f);
  float v = 0.0f, d;
  d = u.x - mean; v += d * d;  d = u.y - mean; v += d * d;
  d = u.z - mean; v += d * d;  d = u.w - mean; v += d * d;
  d = w.x - mean; v += d * d;  d = w.y - mean; v += d * d;
  d = w.z - mean; v += d * d;  d = w.w - mean; v += d * d;
#pragma unroll
  for (int o = 16; o > 0; o >>= 1) v += __shfl_xor(v, o, 32);
  const float rstd = rsqrtf(v * (1.0f / 256.0f) + 1e-5f);
  const int i0 = lane * 4, i1 = 128 + lane * 4;
  float y0 = (u.x - mean) * rstd * g[i0 + 0] + b[i0 + 0];
  float y1 = (u.y - mean) * rstd * g[i0 + 1] + b[i0 + 1];
  float y2 = (u.z - mean) * rstd * g[i0 + 2] + b[i0 + 2];
  float y3 = (u.w - mean) * rstd * g[i0 + 3] + b[i0 + 3];
  float y4 = (w.x - mean) * rstd * g[i1 + 0] + b[i1 + 0];
  float y5 = (w.y - mean) * rstd * g[i1 + 1] + b[i1 + 1];
  float y6 = (w.z - mean) * rstd * g[i1 + 2] + b[i1 + 2];
  float y7 = (w.w - mean) * rstd * g[i1 + 3] + b[i1 + 3];
  float* o32 = out32 + tok * kD;
  bf16_t* o16 = out16 + tok * kD;
  o32[i0 + 0] = y0; o32[i0 + 1] = y1; o32[i0 + 2] = y2; o32[i0 + 3] = y3;
  o32[i1 + 0] = y4; o32[i1 + 1] = y5; o32[i1 + 2] = y6; o32[i1 + 3] = y7;
  o16[i0 + 0] = (bf16_t)y0; o16[i0 + 1] = (bf16_t)y1;
  o16[i0 + 2] = (bf16_t)y2; o16[i0 + 3] = (bf16_t)y3;
  o16[i1 + 0] = (bf16_t)y4; o16[i1 + 1] = (bf16_t)y5;
  o16[i1 + 2] = (bf16_t)y6; o16[i1 + 3] = (bf16_t)y7;
}

// ---------------------------------------------------------------------------
// conv1 (C=16 -> D=256, k=3, SAME) + ReLU, bf16 output. K=48 -> VALU.
// ---------------------------------------------------------------------------
__global__ __launch_bounds__(256)
void conv1_relu_kernel(const float* __restrict__ x, const float* __restrict__ w,
                       const float* __restrict__ bias, bf16_t* __restrict__ out) {
  long i = (long)blockIdx.x * blockDim.x + threadIdx.x;
  if (i >= kBT * (long)kD) return;
  const int o = (int)(i & (kD - 1));
  const long g = i >> 8;
  const int b = (int)(g / kT), t = (int)(g % kT);
  float acc = bias[o];
#pragma unroll
  for (int tap = 0; tap < 3; ++tap) {
    const int tp = t + tap - 1;
    if (tp < 0 || tp >= kT) continue;
    const float* xr = x + ((long)b * kT + tp) * kC;
    const float* wr = w + o * (kC * 3) + tap;
#pragma unroll
    for (int c = 0; c < kC; ++c) acc += xr[c] * wr[c * 3];
  }
  out[i] = (bf16_t)fmaxf(acc, 0.0f);
}

// ---------------------------------------------------------------------------
// Block attention: one workgroup per (batch, 20-token block).
// The contiguous 20x768 bf16 qkv slab is pulled into LDS with
// global_load_async_to_lds_b128 (ASYNCcnt), then 8 heads x 20 queries.
// grid = B*NB, block 256.
// ---------------------------------------------------------------------------
__global__ __launch_bounds__(256)
void block_attn_kernel(const bf16_t* __restrict__ qkv, bf16_t* __restrict__ ctx) {
  __shared__ bf16_t sqkv[kBlk * 3 * kD];  // 30720 B
  const long tok0 = (long)blockIdx.x * kBlk;  // T % BLK == 0: blocks align
  const bf16_t* src = qkv + tok0 * (3 * kD);
  for (int u = threadIdx.x; u < (kBlk * 3 * kD) / 8; u += 256) {
    uint32_t lds = (uint32_t)(uintptr_t)(&sqkv[u * 8]);
    uint64_t g = (uint64_t)(uintptr_t)(src + u * 8);
    asm volatile("global_load_async_to_lds_b128 %0, %1, off"
                 :: "v"(lds), "v"(g) : "memory");
  }
  asm volatile("s_wait_asynccnt 0x0" ::: "memory");
  __syncthreads();
  if (threadIdx.x < kBlk * kH) {
    const int s = threadIdx.x % kBlk;
    const int hh = threadIdx.x / kBlk;
    const bf16_t* q = &sqkv[s * (3 * kD) + hh * kHD];
    float sc[kBlk];
    float mx = -3.4e38f;
    for (int j = 0; j < kBlk; ++j) {
      const bf16_t* kr = &sqkv[j * (3 * kD) + kD + hh * kHD];
      float dd = 0.0f;
#pragma unroll
      for (int e = 0; e < kHD; ++e) dd += (float)q[e] * (float)kr[e];
      dd *= 0.17677669529663687f;  // 1/sqrt(32)
      sc[j] = dd;
      mx = fmaxf(mx, dd);
    }
    float den = 0.0f;
    for (int j = 0; j < kBlk; ++j) { sc[j] = expf(sc[j] - mx); den += sc[j]; }
    const float inv = 1.0f / den;
    bf16_t* outp = ctx + (tok0 + s) * kD + hh * kHD;
    for (int e = 0; e < kHD; ++e) {
      float o = 0.0f;
      for (int j = 0; j < kBlk; ++j)
        o += sc[j] * (float)sqkv[j * (3 * kD) + 2 * kD + hh * kHD + e];
      outp[e] = (bf16_t)(o * inv);
    }
  }
}

// ---------------------------------------------------------------------------
// Final FC on last token: out[b,c] = h[b,T-1,:] . fc_w[c,:] + fc_b[c].
// ---------------------------------------------------------------------------
__global__ __launch_bounds__(512)
void final_fc_kernel(const float* __restrict__ h, const float* __restrict__ w,
                     const float* __restrict__ bias, float* __restrict__ out) {
  const int i = threadIdx.x;
  if (i >= kB * kC) return;
  const int b = i >> 4, c = i & 15;
  const float* row = h + ((long)b * kT + (kT - 1)) * kD;
  const float* wr = w + c * kD;
  float acc = bias[c];
  for (int k = 0; k < kD; ++k) acc += row[k] * wr[k];
  out[i] = acc;
}

// ---------------------------------------------------------------------------
// fp32 -> bf16 conversions.
// ---------------------------------------------------------------------------
__global__ __launch_bounds__(256)
void cvt_f32_bf16(const float* __restrict__ src, bf16_t* __restrict__ dst, long n) {
  long i = (long)blockIdx.x * blockDim.x + threadIdx.x;
  if (i < n) dst[i] = (bf16_t)src[i];
}

// conv2_w (o,i,tap) -> bf16 [o][tap*256 + i]
__global__ __launch_bounds__(256)
void cvt_conv2w(const float* __restrict__ w, bf16_t* __restrict__ dst) {
  const int i = blockIdx.x * blockDim.x + threadIdx.x;
  if (i >= kD * kD * 3) return;
  const int tap = i % 3;
  const int ci = (i / 3) % kD;
  const int o = i / (3 * kD);
  dst[o * 768 + tap * kD + ci] = (bf16_t)w[i];
}

// ===========================================================================
extern "C" void kernel_launch(void* const* d_in, const int* in_sizes, int n_in,
                              void* d_out, int out_size, void* d_ws, size_t ws_size,
                              hipStream_t stream) {
  (void)in_sizes; (void)n_in; (void)out_size; (void)ws_size;
  const float* x          = (const float*)d_in[0];
  const float* conv1_w    = (const float*)d_in[1];
  const float* conv1_b    = (const float*)d_in[2];
  const float* conv2_w    = (const float*)d_in[3];
  const float* conv2_b    = (const float*)d_in[4];
  const float* ln_f_g     = (const float*)d_in[5];
  const float* ln_f_b     = (const float*)d_in[6];
  const float* trend_w    = (const float*)d_in[7];
  const float* trend_b    = (const float*)d_in[8];
  const float* season_w   = (const float*)d_in[9];
  const float* season_b   = (const float*)d_in[10];
  const float* attn_in_w  = (const float*)d_in[11];
  const float* attn_in_b  = (const float*)d_in[12];
  const float* attn_out_w = (const float*)d_in[13];
  const float* attn_out_b = (const float*)d_in[14];
  const float* ln1_g      = (const float*)d_in[15];
  const float* ln1_b      = (const float*)d_in[16];
  const float* ff1_w      = (const float*)d_in[17];
  const float* ff1_b      = (const float*)d_in[18];
  const float* ff2_w      = (const float*)d_in[19];
  const float* ff2_b      = (const float*)d_in[20];
  const float* ln2_g      = (const float*)d_in[21];
  const float* ln2_b      = (const float*)d_in[22];
  const float* fc_w       = (const float*)d_in[23];
  const float* fc_b       = (const float*)d_in[24];
  float* out = (float*)d_out;

  // Workspace carve-out (256-byte aligned slices).
  char* p = (char*)d_ws;
  auto alloc = [&](size_t bytes) -> void* {
    void* r = (void*)p;
    p += (bytes + 255) & ~(size_t)255;
    return r;
  };
  float*  f0     = (float*)alloc(kBT * kD * sizeof(float));   // scratch / h
  float*  f1     = (float*)alloc(kBT * kD * sizeof(float));   // scratch / h
  bf16_t* g0     = (bf16_t*)alloc(kBT * kD * sizeof(bf16_t)); // h bf16
  bf16_t* g1     = (bf16_t*)alloc(kBT * kD * sizeof(bf16_t));
  bf16_t* h1bf   = (bf16_t*)alloc(kBT * kD * sizeof(bf16_t)); // conv1 out
  bf16_t* qkvbf  = (bf16_t*)alloc(kBT * 3 * kD * sizeof(bf16_t));
  bf16_t* ctxbf  = (bf16_t*)alloc(kBT * kD * sizeof(bf16_t));
  bf16_t* hidbf  = (bf16_t*)alloc(kBT * kF * sizeof(bf16_t));
  bf16_t* wC2    = (bf16_t*)alloc((size_t)kD * 768 * sizeof(bf16_t));
  bf16_t* wT     = (bf16_t*)alloc((size_t)kD * kD * sizeof(bf16_t));
  bf16_t* wS     = (bf16_t*)alloc((size_t)kD * kD * sizeof(bf16_t));
  bf16_t* wQKV   = (bf16_t*)alloc((size_t)kL * 3 * kD * kD * sizeof(bf16_t));
  bf16_t* wOut   = (bf16_t*)alloc((size_t)kL * kD * kD * sizeof(bf16_t));
  bf16_t* wFF1   = (bf16_t*)alloc((size_t)kL * kF * kD * sizeof(bf16_t));
  bf16_t* wFF2   = (bf16_t*)alloc((size_t)kL * kD * kF * sizeof(bf16_t));

  auto cvt = [&](const float* src, bf16_t* dst, long n) {
    cvt_f32_bf16<<<dim3((unsigned)((n + 255) / 256)), dim3(256), 0, stream>>>(src, dst, n);
  };

  // Weight conversions (cheap; redone each call for determinism).
  cvt(trend_w, wT, (long)kD * kD);
  cvt(season_w, wS, (long)kD * kD);
  cvt(attn_in_w, wQKV, (long)kL * 3 * kD * kD);
  cvt(attn_out_w, wOut, (long)kL * kD * kD);
  cvt(ff1_w, wFF1, (long)kL * kF * kD);
  cvt(ff2_w, wFF2, (long)kL * kD * kF);
  cvt_conv2w<<<dim3((kD * kD * 3 + 255) / 256), dim3(256), 0, stream>>>(conv2_w, wC2);

  const dim3 blk32(32), blk128(128), blk256(256);
  const unsigned mt16 = (unsigned)(kBT / 16);  // 8000 16-row tiles
  const unsigned mt64 = (unsigned)(kBT / 64);  // 2000 64-row tiles

  // 1) conv1 + relu -> h1bf
  conv1_relu_kernel<<<dim3((unsigned)((kBT * kD + 255) / 256)), blk256, 0, stream>>>(
      x, conv1_w, conv1_b, h1bf);
  // 2) conv2 + relu (WMMA, virtual K=768) -> f0
  conv2_relu_wmma<<<dim3(mt16, 4), blk32, 0, stream>>>(h1bf, wC2, conv2_b, f0);
  // 3) LN_f -> f1 (fp32) + g1 (bf16)
  layernorm_kernel<<<dim3((unsigned)(kBT / 8)), blk256, 0, stream>>>(
      f0, ln_f_g, ln_f_b, f1, g1);
  // 4) h = h + (h@Wt^T + tb) + sin(h@Ws^T + sb)  -> f0 + g0
  trend_season_wmma<<<dim3(mt16, 4), blk32, 0, stream>>>(
      g1, wT, wS, trend_b, season_b, f1, f0, g0);

  // Encoder stack: h fp32 = f0, h bf16 = g0, pre-LN scratch = f1.
  for (int i = 0; i < kL; ++i) {
    // QKV projection: [BT,768] bf16 (TDM-staged weights)
    gemm_bf16_wmma<<<dim3(mt64, 12), blk128, 0, stream>>>(
        g0, wQKV + (size_t)i * 3 * kD * kD, attn_in_b + i * 3 * kD,
        nullptr, nullptr, qkvbf, 3 * kD, kD, 0);
    // Block attention (async-LDS staging) -> ctxbf
    block_attn_kernel<<<dim3((unsigned)(kB * kNB)), blk256, 0, stream>>>(qkvbf, ctxbf);
    // Out projection + residual -> f1 (pre-LN)
    gemm_bf16_wmma<<<dim3(mt64, 4), blk128, 0, stream>>>(
        ctxbf, wOut + (size_t)i * kD * kD, attn_out_b + i * kD,
        f0, f1, nullptr, kD, kD, 0);
    // LN1 -> f0 + g0
    layernorm_kernel<<<dim3((unsigned)(kBT / 8)), blk256, 0, stream>>>(
        f1, ln1_g + i * kD, ln1_b + i * kD, f0, g0);
    // FF1 + relu -> hidbf [BT,1024] bf16
    gemm_bf16_wmma<<<dim3(mt64, 16), blk128, 0, stream>>>(
        g0, wFF1 + (size_t)i * kF * kD, ff1_b + i * kF,
        nullptr, nullptr, hidbf, kF, kD, 1);
    // FF2 + residual -> f1 (pre-LN)
    gemm_bf16_wmma<<<dim3(mt64, 4), blk128, 0, stream>>>(
        hidbf, wFF2 + (size_t)i * kD * kF, ff2_b + i * kD,
        f0, f1, nullptr, kD, kF, 0);
    // LN2 -> f0 + g0
    layernorm_kernel<<<dim3((unsigned)(kBT / 8)), blk256, 0, stream>>>(
        f1, ln2_g + i * kD, ln2_b + i * kD, f0, g0);
  }

  // Final FC on last token -> (B, C)
  final_fc_kernel<<<dim3(1), dim3(512), 0, stream>>>(f0, fc_w, fc_b, out);
}
